// MultiHeadAttention_58377195487667
// MI455X (gfx1250) — matrix operand order repaired and checked by reading, not compiled, not run on hardware
//
#include <hip/hip_runtime.h>
#include <hip/hip_bf16.h>

typedef __attribute__((ext_vector_type(16))) _Float16 v16h;
typedef __attribute__((ext_vector_type(8)))  _Float16 v8h;
typedef __attribute__((ext_vector_type(8)))  float    v8f;
typedef __attribute__((ext_vector_type(4)))  unsigned int u32x4;
typedef __attribute__((ext_vector_type(8)))  int          i32x8;
typedef __attribute__((ext_vector_type(4)))  int          i32x4;

#define B_  4
#define S_  2048
#define D_  768
#define H_  12
#define E_  64
#define BS_ (B_ * S_)          // 8192

// ---------------------------------------------------------------------------
// WMMA helper: D = A(16x32 f16) * B(32x16 f16) + C(16x16 f32)
// ---------------------------------------------------------------------------
__device__ inline v8f wmma_f16(v16h a, v16h b, v8f c) {
  return __builtin_amdgcn_wmma_f32_16x16x32_f16(
      false, a, false, b, (short)0, c, false, false);
}

// ---------------------------------------------------------------------------
// A fragment (16x32 MxK) from a row-major f16 matrix, vectorized:
// lane L -> M = L&15, khi = 16*(L>>4); halves 0..7 = K khi+0..7 (contig 16B),
// halves 8..15 = K khi+8..15 (contig 16B).
// ---------------------------------------------------------------------------
__device__ inline v16h load_a16(const _Float16* __restrict__ base, int ldm,
                                int row, int kb, int lane) {
  const _Float16* p =
      base + (size_t)(row + (lane & 15)) * ldm + kb + ((lane >> 4) * 16);
  v8h lo = *(const v8h*)p;
  v8h hi = *(const v8h*)(p + 8);
  v16h a;
#pragma unroll
  for (int i = 0; i < 8; ++i) { a[i] = lo[i]; a[8 + i] = hi[i]; }
  return a;
}

// B fragment (32x16 KxN) from K-contiguous (transposed) f16 storage:
// source row j = column of the GEMM, contiguous over k.  One 32B load/lane.
__device__ inline v16h load_bt(const _Float16* __restrict__ wt, int ldk,
                               int jbase, int kb, int lane) {
  const _Float16* p =
      wt + (size_t)(jbase + (lane & 15)) * ldk + kb + ((lane >> 4) * 16);
  return *(const v16h*)p;
}

// B fragment for P*V from an LDS tile vt[32][64]: B[k][n] = vt[k][nb+n]
__device__ inline v16h load_vb_lds(const _Float16* __restrict__ vt, int nb,
                                   int lane) {
  const _Float16* p = vt + (size_t)((lane >> 4) * 16) * E_ + nb + (lane & 15);
  v16h b;
#pragma unroll
  for (int h = 0; h < 16; ++h) b[h] = p[(size_t)h * E_];
  return b;
}

// A fragment of P from a 16x32 f16 LDS tile.
__device__ inline v16h load_a_lds(const _Float16* __restrict__ t, int lane) {
  int m   = lane & 15;
  int khi = (lane >> 4) * 16;
  v16h a;
#pragma unroll
  for (int v = 0; v < 8; ++v) {
    int k0 = 2 * v + khi;
    a[2 * v]     = t[m * 32 + k0];
    a[2 * v + 1] = t[m * 32 + k0 + 1];
  }
  return a;
}

// Cross-lane reductions within 16-lane half-groups (match C/D layout).
__device__ inline float half_group_max(float v) {
#pragma unroll
  for (int m = 1; m <= 8; m <<= 1) v = fmaxf(v, __shfl_xor(v, m, 32));
  return v;
}
__device__ inline float half_group_sum(float v) {
#pragma unroll
  for (int m = 1; m <= 8; m <<= 1) v += __shfl_xor(v, m, 32);
  return v;
}

// ---------------------------------------------------------------------------
// Tensor Data Mover: load a (tile_rows x 64) f16 tile, row-stride 64, into
// LDS.  D# packed per ISA 8.3/8.4: count=1, type=2, data_size=1 (2 bytes).
// 6-arg builtin form (clang-23 / therock headers): extra int32x8 group.
// ---------------------------------------------------------------------------
__device__ inline void tdm_load_tile64(unsigned lds_off,
                                       const _Float16* __restrict__ gptr,
                                       unsigned tile_rows,
                                       unsigned tensor_rows) {
  unsigned long long ga = (unsigned long long)(size_t)gptr;
  u32x4 g0;
  g0[0] = 1u;                                        // count=1 (valid user D#)
  g0[1] = lds_off;                                   // lds_addr (bytes)
  g0[2] = (unsigned)(ga & 0xffffffffu);              // global_addr[31:0]
  g0[3] = (unsigned)((ga >> 32) & 0x01ffffffu)       // global_addr[56:32]
        | (2u << 30);                                // type=2 ("image")
  i32x8 g1;
  g1[0] = (int)(1u << 16);            // data_size=1 (2B); mask/flags = 0
  g1[1] = (int)(64u << 16);           // tensor_dim0[15:0] = 64
  g1[2] = (int)(tensor_rows << 16);   // dim0 hi=0 | tensor_dim1[15:0]
  g1[3] = (int)(64u << 16);           // dim1 hi=0 | tile_dim0 = 64
  g1[4] = (int)tile_rows;             // tile_dim1 | tile_dim2=0
  g1[5] = 64;                         // tensor_dim0_stride = 64 elements
  g1[6] = 0;
  g1[7] = 0;
  i32x4 z4 = {0, 0, 0, 0};            // groups 2/3: tile_dim3/4 = 0 (unused)
  i32x8 z8 = {0, 0, 0, 0, 0, 0, 0, 0};
  __builtin_amdgcn_tensor_load_to_lds(g0, g1, z4, z4, z8, 0);
}

// ---------------------------------------------------------------------------
// Pre-pass: x -> f16 (row-major)
// ---------------------------------------------------------------------------
__global__ __launch_bounds__(256) void cvt_x_kernel(
    const float* __restrict__ x, _Float16* __restrict__ xh) {
  size_t i = ((size_t)blockIdx.x * 256 + threadIdx.x) * 4;
  const float4 v = *(const float4*)(x + i);
  xh[i + 0] = (_Float16)v.x;
  xh[i + 1] = (_Float16)v.y;
  xh[i + 2] = (_Float16)v.z;
  xh[i + 3] = (_Float16)v.w;
}

// Pre-pass: Wq/Wk/Wv [H][D][64] -> wt[2304][768] f16, wt[j][d], j = sel*768+h*64+e
__global__ __launch_bounds__(256) void cvt_wqkv_kernel(
    const float* __restrict__ Wq, const float* __restrict__ Wk,
    const float* __restrict__ Wv, _Float16* __restrict__ wt) {
  size_t gid = (size_t)blockIdx.x * 256 + threadIdx.x;
  int j  = (int)(gid / (D_ / 4));
  int d0 = (int)(gid % (D_ / 4)) * 4;
  int sel = j / D_;
  int jj  = j % D_;
  int h   = jj / E_;
  int e   = jj % E_;
  const float* W = (sel == 0 ? Wq : (sel == 1 ? Wk : Wv));
  const float* src = W + (size_t)h * D_ * E_ + (size_t)d0 * E_ + e;
  _Float16* dst = wt + (size_t)j * D_ + d0;
#pragma unroll
  for (int d = 0; d < 4; ++d) dst[d] = (_Float16)src[(size_t)d * E_];
}

// Pre-pass: Wo [768][768] -> wot[j][d] = Wo[d][j] f16
__global__ __launch_bounds__(256) void cvt_wo_kernel(
    const float* __restrict__ Wo, _Float16* __restrict__ wot) {
  size_t gid = (size_t)blockIdx.x * 256 + threadIdx.x;
  int j  = (int)(gid / (D_ / 4));
  int d0 = (int)(gid % (D_ / 4)) * 4;
  _Float16* dst = wot + (size_t)j * D_ + d0;
#pragma unroll
  for (int d = 0; d < 4; ++d) dst[d] = (_Float16)Wo[(size_t)(d0 + d) * D_ + j];
}

// ---------------------------------------------------------------------------
// Kernel 1: fused QKV projection, 16x64 tile per wave (A reused over 4 B).
// Logical GEMM [8192 x 768] x [768 x 2304] -> qkv f16 [3][B][H][S][64].
// ---------------------------------------------------------------------------
__global__ __launch_bounds__(256) void qkv_proj_kernel(
    const _Float16* __restrict__ xh, const _Float16* __restrict__ wt,
    const float* __restrict__ bq, const float* __restrict__ bk,
    const float* __restrict__ bv, _Float16* __restrict__ qkv) {
  int lane = threadIdx.x & 31;
  int wid  = threadIdx.x >> 5;
  int gw   = blockIdx.x * 8 + wid;
  const int TM = BS_ / 16;          // 512 row tiles
  int tm = gw % TM;
  int tg = gw / TM;                 // 0..35 groups of 64 columns
  int jb64 = tg * 64;
  int sel  = jb64 / D_;
  int h    = (jb64 % D_) / E_;
  int row  = tm * 16;
  const float* bias = (sel == 0 ? bq : (sel == 1 ? bk : bv)) + h * E_;

  v8f acc0 = {}, acc1 = {}, acc2 = {}, acc3 = {};
#pragma unroll 2
  for (int kb = 0; kb < D_; kb += 32) {
    __builtin_prefetch(xh + (size_t)(row + (lane & 15)) * D_ + kb + 128, 0, 0);
    v16h a = load_a16(xh, D_, row, kb, lane);
    acc0 = wmma_f16(a, load_bt(wt, D_, jb64 +  0, kb, lane), acc0);
    acc1 = wmma_f16(a, load_bt(wt, D_, jb64 + 16, kb, lane), acc1);
    acc2 = wmma_f16(a, load_bt(wt, D_, jb64 + 32, kb, lane), acc2);
    acc3 = wmma_f16(a, load_bt(wt, D_, jb64 + 48, kb, lane), acc3);
  }

  int n   = lane & 15;
  int mhi = (lane >> 4) * 8;
  float bc0 = bias[0 + n], bc1 = bias[16 + n], bc2 = bias[32 + n],
        bc3 = bias[48 + n];
#pragma unroll
  for (int r = 0; r < 8; ++r) {
    int m = row + r + mhi;
    int bb = m / S_;
    int ss = m % S_;
    size_t base = ((((size_t)sel * B_ + bb) * H_ + h) * S_ + ss) * E_;
    qkv[base +  0 + n] = (_Float16)(acc0[r] + bc0);
    qkv[base + 16 + n] = (_Float16)(acc1[r] + bc1);
    qkv[base + 32 + n] = (_Float16)(acc2[r] + bc2);
    qkv[base + 48 + n] = (_Float16)(acc3[r] + bc3);
  }
}

// ---------------------------------------------------------------------------
// Kernel 2: flash attention.  8 waves/block share one (b,h); K/V 32x64 tiles
// staged into LDS once per block by the Tensor Data Mover; each wave handles
// its own 16-query tile with online softmax.  8 WMMAs per wave per KV step.
// ---------------------------------------------------------------------------
__global__ __launch_bounds__(256) void attn_kernel(
    const _Float16* __restrict__ qkv, _Float16* __restrict__ ao) {
  __shared__ _Float16 kt[32 * 64];       // K tile  (4 KB)
  __shared__ _Float16 vt[32 * 64];       // V tile  (4 KB)
  __shared__ _Float16 plds[8][16 * 32];  // per-wave P tiles (8 KB)

  int lane = threadIdx.x & 31;
  int wid  = threadIdx.x >> 5;
  int bh = blockIdx.x / 16;              // (b,h) of this block
  int qg = blockIdx.x % 16;              // query-tile group
  int b = bh / H_;
  int h = bh % H_;
  int qt = qg * 8 + wid;                 // this wave's 16-row query tile

  const size_t head = (((size_t)b * H_ + h) * S_) * E_;
  const size_t pln  = (size_t)B_ * H_ * S_ * E_;
  const _Float16* Qh = qkv + head;
  const _Float16* Kh = qkv + pln + head;
  const _Float16* Vh = qkv + 2 * pln + head;

  unsigned kt_off = (unsigned)(size_t)(&kt[0]);   // LDS byte offset
  unsigned vt_off = (unsigned)(size_t)(&vt[0]);

  v16h aq0 = load_a16(Qh, E_, qt * 16, 0, lane);
  v16h aq1 = load_a16(Qh, E_, qt * 16, 32, lane);

  v8f o0 = {}, o1 = {}, o2 = {}, o3 = {};
  float mrow[8], lrow[8];
#pragma unroll
  for (int r = 0; r < 8; ++r) { mrow[r] = -__builtin_inff(); lrow[r] = 0.f; }
  const float scale = 0.125f;  // 1/sqrt(64)

  for (int j = 0; j < S_; j += 32) {
    __syncthreads();  // all readers of previous tile are done
    if (wid == 0) {
      tdm_load_tile64(kt_off, Kh + (size_t)j * E_, 32u, (unsigned)S_);
      tdm_load_tile64(vt_off, Vh + (size_t)j * E_, 32u, (unsigned)S_);
      __builtin_amdgcn_s_wait_tensorcnt(0);
    }
    __syncthreads();  // tiles visible to all waves

    // ---- scores: S = Q * K^T ; B[k][n] = kt[n][kb+k], contiguous per lane
    v8f s0 = {}, s1 = {};
    s0 = wmma_f16(aq0, load_bt(kt, E_,  0,  0, lane), s0);
    s0 = wmma_f16(aq1, load_bt(kt, E_,  0, 32, lane), s0);
    s1 = wmma_f16(aq0, load_bt(kt, E_, 16,  0, lane), s1);
    s1 = wmma_f16(aq1, load_bt(kt, E_, 16, 32, lane), s1);

    // ---- online softmax (VGPR r <-> row; 16-lane group shares a row)
    float corr[8];
#pragma unroll
    for (int r = 0; r < 8; ++r) {
      float v0 = s0[r] * scale;
      float v1 = s1[r] * scale;
      float mx = half_group_max(fmaxf(v0, v1));
      float mn = fmaxf(mrow[r], mx);
      float c  = __expf(mrow[r] - mn);
      float p0 = __expf(v0 - mn);
      float p1 = __expf(v1 - mn);
      lrow[r] = lrow[r] * c + half_group_sum(p0 + p1);
      mrow[r] = mn;
      corr[r] = c;
      s0[r] = p0;
      s1[r] = p1;
    }
#pragma unroll
    for (int r = 0; r < 8; ++r) {
      o0[r] *= corr[r]; o1[r] *= corr[r]; o2[r] *= corr[r]; o3[r] *= corr[r];
    }

    // ---- re-layout P: D-fragment -> LDS -> A-fragment (per-wave region)
    {
      int n   = lane & 15;
      int mhi = (lane >> 4) * 8;
#pragma unroll
      for (int r = 0; r < 8; ++r) {
        plds[wid][(r + mhi) * 32 + n]      = (_Float16)s0[r];
        plds[wid][(r + mhi) * 32 + 16 + n] = (_Float16)s1[r];
      }
    }
    __syncthreads();
    v16h ap = load_a_lds(&plds[wid][0], lane);

    // ---- O += P * V (four 16-wide head-dim slices from the LDS V tile)
    o0 = wmma_f16(ap, load_vb_lds(vt,  0, lane), o0);
    o1 = wmma_f16(ap, load_vb_lds(vt, 16, lane), o1);
    o2 = wmma_f16(ap, load_vb_lds(vt, 32, lane), o2);
    o3 = wmma_f16(ap, load_vb_lds(vt, 48, lane), o3);
  }

  // ---- normalize and store concat-head layout ao[b][s][h*64 + e] (f16)
  int n   = lane & 15;
  int mhi = (lane >> 4) * 8;
#pragma unroll
  for (int r = 0; r < 8; ++r) {
    float inv = 1.f / lrow[r];
    int m = qt * 16 + r + mhi;
    size_t base = ((size_t)b * S_ + m) * D_ + h * E_;
    ao[base +  0 + n] = (_Float16)(o0[r] * inv);
    ao[base + 16 + n] = (_Float16)(o1[r] * inv);
    ao[base + 32 + n] = (_Float16)(o2[r] * inv);
    ao[base + 48 + n] = (_Float16)(o3[r] * inv);
  }
}

// ---------------------------------------------------------------------------
// Kernel 3: output projection [8192 x 768] x [768 x 768] + bias -> f32 out.
// 16x64 tile per wave; B from pre-transposed f16 Wo.
// ---------------------------------------------------------------------------
__global__ __launch_bounds__(256) void out_proj_kernel(
    const _Float16* __restrict__ ao, const _Float16* __restrict__ wot,
    const float* __restrict__ bo, float* __restrict__ out) {
  int lane = threadIdx.x & 31;
  int wid  = threadIdx.x >> 5;
  int gw   = blockIdx.x * 8 + wid;
  const int TM = BS_ / 16;          // 512
  int tm = gw % TM;
  int tg = gw / TM;                 // 0..11
  int jb64 = tg * 64;
  int row  = tm * 16;

  v8f acc0 = {}, acc1 = {}, acc2 = {}, acc3 = {};
#pragma unroll 2
  for (int kb = 0; kb < D_; kb += 32) {
    __builtin_prefetch(ao + (size_t)(row + (lane & 15)) * D_ + kb + 128, 0, 0);
    v16h a = load_a16(ao, D_, row, kb, lane);
    acc0 = wmma_f16(a, load_bt(wot, D_, jb64 +  0, kb, lane), acc0);
    acc1 = wmma_f16(a, load_bt(wot, D_, jb64 + 16, kb, lane), acc1);
    acc2 = wmma_f16(a, load_bt(wot, D_, jb64 + 32, kb, lane), acc2);
    acc3 = wmma_f16(a, load_bt(wot, D_, jb64 + 48, kb, lane), acc3);
  }

  int n   = lane & 15;
  int mhi = (lane >> 4) * 8;
  float bc0 = bo[jb64 + 0 + n],  bc1 = bo[jb64 + 16 + n],
        bc2 = bo[jb64 + 32 + n], bc3 = bo[jb64 + 48 + n];
#pragma unroll
  for (int r = 0; r < 8; ++r) {
    int m = row + r + mhi;
    float* o = out + (size_t)m * D_ + jb64;
    o[ 0 + n] = acc0[r] + bc0;
    o[16 + n] = acc1[r] + bc1;
    o[32 + n] = acc2[r] + bc2;
    o[48 + n] = acc3[r] + bc3;
  }
}

// ---------------------------------------------------------------------------
// Launch
// ---------------------------------------------------------------------------
extern "C" void kernel_launch(void* const* d_in, const int* in_sizes, int n_in,
                              void* d_out, int out_size, void* d_ws,
                              size_t ws_size, hipStream_t stream) {
  (void)in_sizes; (void)n_in; (void)out_size; (void)ws_size;
  const float* x  = (const float*)d_in[0];
  const float* Wq = (const float*)d_in[1];
  const float* bq = (const float*)d_in[2];
  const float* Wk = (const float*)d_in[3];
  const float* bk = (const float*)d_in[4];
  const float* Wv = (const float*)d_in[5];
  const float* bv = (const float*)d_in[6];
  const float* Wo = (const float*)d_in[7];
  const float* bo = (const float*)d_in[8];

  _Float16* qkv = (_Float16*)d_ws;                        // [3][B][H][S][64]
  _Float16* ao  = qkv + (size_t)3 * B_ * H_ * S_ * E_;    // [B][S][768]
  _Float16* xh  = ao + (size_t)BS_ * D_;                  // [8192][768]
  _Float16* wtq = xh + (size_t)BS_ * D_;                  // [2304][768]
  _Float16* wto = wtq + (size_t)3 * D_ * D_;              // [768][768]

  cvt_x_kernel<<<(BS_ * D_ / 4) / 256, 256, 0, stream>>>(x, xh);
  cvt_wqkv_kernel<<<(3 * D_ * D_ / 4) / 256, 256, 0, stream>>>(Wq, Wk, Wv, wtq);
  cvt_wo_kernel<<<(D_ * D_ / 4) / 256, 256, 0, stream>>>(Wo, wto);

  // K1: 512 row-tiles * 36 col-groups / 8 waves per block
  qkv_proj_kernel<<<(512 * 36) / 8, 256, 0, stream>>>(xh, wtq, bq, bk, bv, qkv);
  // K2: B*H blocks * 16 query-groups
  attn_kernel<<<B_ * H_ * 16, 256, 0, stream>>>(qkv, ao);
  // K3: 512 row-tiles * 12 col-groups / 8 waves per block
  out_proj_kernel<<<(512 * 12) / 8, 256, 0, stream>>>(ao, wto, bo,
                                                      (float*)d_out);
}
